// MambaBlock_481036337187
// MI455X (gfx1250) — compile-verified, matrix operand order checked by reading, not run on hardware
//
#include <hip/hip_runtime.h>
#include <hip/hip_bf16.h>

// ---------------------------------------------------------------------------
// MambaBlock for MI455X (gfx1250): LN -> dual bf16-WMMA GEMM -> chunked scan
// -> readout -> bf16-WMMA GEMM.  wave32, V_WMMA_F32_16X16X32_BF16.
// Register-blocked GEMMs: wave = 64(M) x 16(N) against 2 B-fragments
// (8 WMMAs per 6 fragment loads); all of a k-step's fragments are loaded
// into distinct locals before the WMMA burst so waits can be graduated.
// ---------------------------------------------------------------------------

typedef __attribute__((ext_vector_type(16))) __bf16 v16bf;
typedef __attribute__((ext_vector_type(8)))  float  v8f;
typedef __attribute__((ext_vector_type(4)))  unsigned int u32x4;

static constexpr int BB = 8;
static constexpr int L  = 4096;
static constexpr int D  = 512;
static constexpr int H  = 512;
static constexpr int M  = BB * L;          // 32768 rows
static constexpr int CH = 64;              // scan chunks per sequence
static constexpr int LC = L / CH;          // 64 steps per chunk

// float -> bf16, round to nearest even
__device__ inline unsigned short f2bf(float f) {
  unsigned int u = __builtin_bit_cast(unsigned int, f);
  u += 0x7FFFu + ((u >> 16) & 1u);
  return (unsigned short)(u >> 16);
}

// Load one 16x32 bf16 WMMA fragment slice for this lane.
// p0 already includes row*ld + k0 + (lane>=16 ? 8 : 0).
// VGPR0-3 <- K[0..7] (or 8..15), VGPR4-7 <- K[16..23] (or 24..31).
__device__ inline v16bf load_frag(const unsigned short* p0) {
  union { v16bf v; u32x4 u[2]; } f;
  f.u[0] = *(const u32x4*)(p0);
  f.u[1] = *(const u32x4*)(p0 + 16);
  return f.v;
}

// ---------------------------------------------------------------------------
// 0) weight transpose + bf16: W[D,H] (row-major) -> Wt[H,D] (row-major)
// ---------------------------------------------------------------------------
__global__ void __launch_bounds__(256) wtrans_kernel(
    const float* __restrict__ W, unsigned short* __restrict__ Wt) {
  int tid = blockIdx.x * 256 + threadIdx.x;      // over D*H
  int d = tid % D;                               // fast index -> coalesced write
  int h = tid / D;
  if (h < H) Wt[h * D + d] = f2bf(W[d * H + h]);
}

// ---------------------------------------------------------------------------
// 1) LayerNorm over channels, emit bf16.  One wave per row.
// ---------------------------------------------------------------------------
__global__ void __launch_bounds__(256) ln_kernel(
    const float* __restrict__ x, const float* __restrict__ gamma,
    const float* __restrict__ beta, unsigned short* __restrict__ xn) {
  const int wave = threadIdx.x >> 5;
  const int lane = threadIdx.x & 31;
  const int row  = blockIdx.x * 8 + wave;
  const float* px = x + (size_t)row * D;

  float v[16];
  float s = 0.f, s2 = 0.f;
  #pragma unroll
  for (int i = 0; i < 16; ++i) {
    float t = px[i * 32 + lane];
    v[i] = t; s += t; s2 += t * t;
  }
  #pragma unroll
  for (int off = 16; off; off >>= 1) {
    s  += __shfl_xor(s,  off, 32);
    s2 += __shfl_xor(s2, off, 32);
  }
  const float mu  = s * (1.f / (float)D);
  const float var = s2 * (1.f / (float)D) - mu * mu;
  const float rs  = rsqrtf(var + 1e-5f);

  unsigned short* po = xn + (size_t)row * D;
  #pragma unroll
  for (int i = 0; i < 16; ++i) {
    int d = i * 32 + lane;
    po[d] = f2bf((v[i] - mu) * rs * gamma[d] + beta[d]);
  }
}

// ---------------------------------------------------------------------------
// 2) Fused dual GEMM: Bt = xn*W_B + b_B,  Ct = xn*W_C + b_C
//    Wave tile: 64(M) x 16(N), both outputs.  4 A frags + 2 W frags loaded
//    up-front per k-step, then an 8-WMMA burst.  Waves in a block share the
//    same 64-row A strip (WGP$ reuse across the 8 N-tiles).
// ---------------------------------------------------------------------------
__global__ void __launch_bounds__(256) gemm_bc_kernel(
    const unsigned short* __restrict__ Xn,
    const unsigned short* __restrict__ WBt,
    const unsigned short* __restrict__ WCt,
    const float* __restrict__ bB, const float* __restrict__ bC,
    float* __restrict__ Bt, float* __restrict__ Ct) {
  const int wave = threadIdx.x >> 5;
  const int lane = threadIdx.x & 31;
  const int m0 = blockIdx.x * 64;                 // 64-row strip per block
  const int n0 = blockIdx.y * 128 + wave * 16;    // wave picks its N tile
  const int r  = lane & 15;
  const int hi = lane >> 4;

  const unsigned short* ap  = Xn  + (size_t)(m0 + r) * D + hi * 8;
  const unsigned short* bpB = WBt + (size_t)(n0 + r) * D + hi * 8;
  const unsigned short* bpC = WCt + (size_t)(n0 + r) * D + hi * 8;

  v8f accB[4] = {{}, {}, {}, {}};
  v8f accC[4] = {{}, {}, {}, {}};
  #pragma unroll 2
  for (int k0 = 0; k0 < D; k0 += 32) {
    // ---- load phase: 6 independent fragments (12 b128 loads) ----
    v16bf a0 = load_frag(ap + 0 * 16 * D + k0);
    v16bf a1 = load_frag(ap + 1 * 16 * D + k0);
    v16bf a2 = load_frag(ap + 2 * 16 * D + k0);
    v16bf a3 = load_frag(ap + 3 * 16 * D + k0);
    v16bf wb = load_frag(bpB + k0);
    v16bf wc = load_frag(bpC + k0);
    __builtin_prefetch(ap + k0 + 32, 0, 0);       // global_prefetch_b8
    // ---- compute phase: 8 WMMAs ----
    accB[0] = __builtin_amdgcn_wmma_f32_16x16x32_bf16(
        false, a0, false, wb, (short)0, accB[0], false, false);
    accC[0] = __builtin_amdgcn_wmma_f32_16x16x32_bf16(
        false, a0, false, wc, (short)0, accC[0], false, false);
    accB[1] = __builtin_amdgcn_wmma_f32_16x16x32_bf16(
        false, a1, false, wb, (short)0, accB[1], false, false);
    accC[1] = __builtin_amdgcn_wmma_f32_16x16x32_bf16(
        false, a1, false, wc, (short)0, accC[1], false, false);
    accB[2] = __builtin_amdgcn_wmma_f32_16x16x32_bf16(
        false, a2, false, wb, (short)0, accB[2], false, false);
    accC[2] = __builtin_amdgcn_wmma_f32_16x16x32_bf16(
        false, a2, false, wc, (short)0, accC[2], false, false);
    accB[3] = __builtin_amdgcn_wmma_f32_16x16x32_bf16(
        false, a3, false, wb, (short)0, accB[3], false, false);
    accC[3] = __builtin_amdgcn_wmma_f32_16x16x32_bf16(
        false, a3, false, wc, (short)0, accC[3], false, false);
  }
  const int col = n0 + r;
  const float vb = bB[col], vc = bC[col];
  #pragma unroll
  for (int mi = 0; mi < 4; ++mi) {
    const int mbase = m0 + mi * 16 + hi * 8;      // VGPR i <-> row mbase+i
    #pragma unroll
    for (int i = 0; i < 8; ++i) {
      Bt[(size_t)(mbase + i) * H + col] = accB[mi][i] + vb;
      Ct[(size_t)(mbase + i) * H + col] = accC[mi][i] + vc;
    }
  }
}

// ---------------------------------------------------------------------------
// 3) Scan pass 1: per (b,chunk,h) local scan with zero init, in place on Bt.
// ---------------------------------------------------------------------------
__global__ void __launch_bounds__(256) scan_local_kernel(
    float* __restrict__ Bt, const float* __restrict__ A,
    float* __restrict__ carry) {
  int tid = blockIdx.x * 256 + threadIdx.x;    // BB*CH*H threads, h fastest
  int h = tid % H;
  int c = (tid / H) % CH;
  int b = tid / (H * CH);
  const float a = A[h];
  float hs = 0.f;
  float* p = Bt + ((size_t)(b * L + c * LC)) * H + h;
  for (int t = 0; t < LC; ++t) {
    hs = a * hs + p[(size_t)t * H];
    p[(size_t)t * H] = hs;
  }
  carry[(size_t)(b * CH + c) * H + h] = hs;    // chunk's local final state
}

// ---------------------------------------------------------------------------
// 4) Scan pass 2: inter-chunk scan; rewrite carry[] with each chunk's carry-IN.
// ---------------------------------------------------------------------------
__global__ void __launch_bounds__(256) scan_chunk_kernel(
    float* __restrict__ carry, const float* __restrict__ A) {
  int tid = blockIdx.x * 256 + threadIdx.x;    // BB*H threads
  int h = tid % H;
  int b = tid / H;
  const float a = A[h];
  float aLC = 1.f;
  for (int i = 0; i < LC; ++i) aLC *= a;       // a^LC (constant per channel)
  float run = 0.f;
  for (int c = 0; c < CH; ++c) {
    size_t i = (size_t)(b * CH + c) * H + h;
    float lf = carry[i];
    carry[i] = run;                            // carry-in for chunk c
    run = aLC * run + lf;
  }
}

// ---------------------------------------------------------------------------
// 5) Scan pass 3: h_t = local_h_t + a^(t+1)*carry_in;  y = Ct * h_t  (bf16)
// ---------------------------------------------------------------------------
__global__ void __launch_bounds__(256) scan_fix_kernel(
    const float* __restrict__ Hl, const float* __restrict__ Ct,
    const float* __restrict__ carry, const float* __restrict__ A,
    unsigned short* __restrict__ Y) {
  int tid = blockIdx.x * 256 + threadIdx.x;    // BB*CH*H threads, h fastest
  int h = tid % H;
  int c = (tid / H) % CH;
  int b = tid / (H * CH);
  const float a   = A[h];
  const float cin = carry[(size_t)(b * CH + c) * H + h];
  size_t base = ((size_t)(b * L + c * LC)) * H + h;
  float pw = a;
  for (int t = 0; t < LC; ++t) {
    float ht = Hl[base + (size_t)t * H] + pw * cin;
    float y  = ht * Ct[base + (size_t)t * H];
    Y[base + (size_t)t * H] = f2bf(y);
    pw *= a;
  }
}

// ---------------------------------------------------------------------------
// 6) Output GEMM: out = Y * W_out + b_out   (f32 result)
//    Wave tile: 64(M) x 32(N): 4 A frags + 2 W frags -> 8 WMMAs per k-step.
// ---------------------------------------------------------------------------
__global__ void __launch_bounds__(256) gemm_out_kernel(
    const unsigned short* __restrict__ Y,
    const unsigned short* __restrict__ WOt,
    const float* __restrict__ bO, float* __restrict__ Out) {
  const int wave = threadIdx.x >> 5;
  const int lane = threadIdx.x & 31;
  const int m0 = blockIdx.x * 64;
  const int n0 = (blockIdx.y * 8 + wave) * 32;   // wave covers cols n0..n0+31
  const int r  = lane & 15;
  const int hi = lane >> 4;

  const unsigned short* ap  = Y   + (size_t)(m0 + r) * H + hi * 8;
  const unsigned short* bp0 = WOt + (size_t)(n0 + r) * H + hi * 8;
  const unsigned short* bp1 = WOt + (size_t)(n0 + 16 + r) * H + hi * 8;

  v8f acc0[4] = {{}, {}, {}, {}};
  v8f acc1[4] = {{}, {}, {}, {}};
  #pragma unroll 2
  for (int k0 = 0; k0 < H; k0 += 32) {
    v16bf a0 = load_frag(ap + 0 * 16 * H + k0);
    v16bf a1 = load_frag(ap + 1 * 16 * H + k0);
    v16bf a2 = load_frag(ap + 2 * 16 * H + k0);
    v16bf a3 = load_frag(ap + 3 * 16 * H + k0);
    v16bf w0 = load_frag(bp0 + k0);
    v16bf w1 = load_frag(bp1 + k0);
    __builtin_prefetch(ap + k0 + 32, 0, 0);
    acc0[0] = __builtin_amdgcn_wmma_f32_16x16x32_bf16(
        false, a0, false, w0, (short)0, acc0[0], false, false);
    acc1[0] = __builtin_amdgcn_wmma_f32_16x16x32_bf16(
        false, a0, false, w1, (short)0, acc1[0], false, false);
    acc0[1] = __builtin_amdgcn_wmma_f32_16x16x32_bf16(
        false, a1, false, w0, (short)0, acc0[1], false, false);
    acc1[1] = __builtin_amdgcn_wmma_f32_16x16x32_bf16(
        false, a1, false, w1, (short)0, acc1[1], false, false);
    acc0[2] = __builtin_amdgcn_wmma_f32_16x16x32_bf16(
        false, a2, false, w0, (short)0, acc0[2], false, false);
    acc1[2] = __builtin_amdgcn_wmma_f32_16x16x32_bf16(
        false, a2, false, w1, (short)0, acc1[2], false, false);
    acc0[3] = __builtin_amdgcn_wmma_f32_16x16x32_bf16(
        false, a3, false, w0, (short)0, acc0[3], false, false);
    acc1[3] = __builtin_amdgcn_wmma_f32_16x16x32_bf16(
        false, a3, false, w1, (short)0, acc1[3], false, false);
  }
  const int c0 = n0 + r, c1 = n0 + 16 + r;
  const float vb0 = bO[c0], vb1 = bO[c1];
  #pragma unroll
  for (int mi = 0; mi < 4; ++mi) {
    const int mbase = m0 + mi * 16 + hi * 8;
    #pragma unroll
    for (int i = 0; i < 8; ++i) {
      Out[(size_t)(mbase + i) * H + c0] = acc0[mi][i] + vb0;
      Out[(size_t)(mbase + i) * H + c1] = acc1[mi][i] + vb1;
    }
  }
}

// ---------------------------------------------------------------------------
extern "C" void kernel_launch(void* const* d_in, const int* in_sizes, int n_in,
                              void* d_out, int out_size, void* d_ws, size_t ws_size,
                              hipStream_t stream) {
  const float* x     = (const float*)d_in[0];
  const float* gamma = (const float*)d_in[1];
  const float* beta  = (const float*)d_in[2];
  const float* W_B   = (const float*)d_in[3];
  const float* b_B   = (const float*)d_in[4];
  const float* W_C   = (const float*)d_in[5];
  const float* b_C   = (const float*)d_in[6];
  const float* W_out = (const float*)d_in[7];
  const float* b_out = (const float*)d_in[8];
  const float* A     = (const float*)d_in[9];
  float* out = (float*)d_out;

  // workspace carve-out (256B aligned)
  char* ws = (char*)d_ws;
  size_t off = 0;
  auto carve = [&](size_t bytes) {
    void* p = ws + off;
    off += (bytes + 255) & ~(size_t)255;
    return p;
  };
  unsigned short* xnY  = (unsigned short*)carve((size_t)M * D * 2); // xn, later reused as Y
  unsigned short* WBt  = (unsigned short*)carve((size_t)H * D * 2);
  unsigned short* WCt  = (unsigned short*)carve((size_t)H * D * 2);
  unsigned short* WOt  = (unsigned short*)carve((size_t)H * H * 2);
  float*          Bt   = (float*)carve((size_t)M * H * 4);
  float*          Ct   = (float*)carve((size_t)M * H * 4);
  float*          cry  = (float*)carve((size_t)BB * CH * H * 4);
  (void)ws_size; (void)in_sizes; (void)n_in; (void)out_size;

  // 0) weight transposes (bf16)
  {
    int n = D * H, blocks = (n + 255) / 256;
    wtrans_kernel<<<blocks, 256, 0, stream>>>(W_B,   WBt);
    wtrans_kernel<<<blocks, 256, 0, stream>>>(W_C,   WCt);
    wtrans_kernel<<<blocks, 256, 0, stream>>>(W_out, WOt);
  }
  // 1) LayerNorm -> bf16 xn
  ln_kernel<<<M / 8, 256, 0, stream>>>(x, gamma, beta, xnY);
  // 2) dual WMMA GEMM -> Bt, Ct   (grid: 512 M-strips x 4 N-groups)
  gemm_bc_kernel<<<dim3(M / 64, H / 128), 256, 0, stream>>>(
      xnY, WBt, WCt, b_B, b_C, Bt, Ct);
  // 3-5) chunked linear recurrence + readout -> Y (bf16, reuses xn buffer)
  scan_local_kernel<<<(BB * CH * H) / 256, 256, 0, stream>>>(Bt, A, cry);
  scan_chunk_kernel<<<(BB * H) / 256, 256, 0, stream>>>(cry, A);
  scan_fix_kernel<<<(BB * CH * H) / 256, 256, 0, stream>>>(Bt, Ct, cry, A, xnY);
  // 6) output WMMA GEMM -> d_out  (grid: 512 M-strips x 2 N-groups)
  gemm_out_kernel<<<dim3(M / 64, H / (32 * 8)), 256, 0, stream>>>(
      xnY, WOt, b_out, out);
}
